// NeuralRDE_28707561406834
// MI455X (gfx1250) — compile-verified, matrix operand order checked by reading, not run on hardware
//
#include <hip/hip_runtime.h>
#include <hip/hip_bf16.h>

// ---------------------------------------------------------------------------
// Neural RDE on MI455X (gfx1250).
//
// Math optimization: the huge (512x256)x(256x7936) matmul contracted with a
// 62-vector "seg" collapses to 32 precomputed effective matrices
//   W_eff[i] (128x256) = sum_l W_m[h*62+l, :] * logsig[i, l+1] / dInterval[i]
//   b_eff[i] (128)     = sum_l b_m[h*62+l]    * logsig[i, l+1] / dInterval[i]
// so each vector-field eval is three small GEMMs, all run on
// v_wmma_f32_16x16x32_f16 (f16 in, f32 accumulate).
// ---------------------------------------------------------------------------

typedef __attribute__((ext_vector_type(16))) _Float16 v16h;
typedef __attribute__((ext_vector_type(8)))  _Float16 v8h;
typedef __attribute__((ext_vector_type(8)))  float    v8f;

#define B_   512
#define H_   128
#define VF_  256
#define L_   62
#define NSEG 32
#define NSTEPS 20

// Load a 16x32 f16 WMMA A/B fragment from a row-major (rows x K) matrix.
// `base` points at element (row0, k0).  Per CDNA5 ISA 16-bit A layout:
//   lanes 0-15 : row = lane,    K = {k0+0..7,  k0+16..23}
//   lanes 16-31: row = lane-16, K = {k0+8..15, k0+24..31}
// (B fragments from a (N x K) weight use the identical pattern: column = lane.)
__device__ __forceinline__ v16h load_frag(const _Float16* __restrict__ base, int ld) {
    int lane = threadIdx.x & 31;
    const _Float16* p = base + (size_t)(lane & 15) * ld + ((lane >> 4) << 3);
    v8h lo = *(const v8h*)(p);
    v8h hi = *(const v8h*)(p + 16);
    return __builtin_shufflevector(lo, hi, 0,1,2,3,4,5,6,7,8,9,10,11,12,13,14,15);
}

// ---------------- precompute kernels ----------------

// W_eff: one block per (seg i, output row h), 256 threads over VF columns.
__global__ __launch_bounds__(256)
void pk_weff(const float* __restrict__ Wm, const float* __restrict__ logsig,
             const float* __restrict__ intervals, _Float16* __restrict__ Weff) {
    int v = threadIdx.x;           // 0..255
    int h = blockIdx.x & (H_ - 1); // 0..127
    int i = blockIdx.x >> 7;       // 0..31
    float scale = 1.0f / (intervals[i + 1] - intervals[i]);
    const float* ls = logsig + i * 63;
    float acc = 0.0f;
    for (int l = 0; l < L_; ++l)
        acc += Wm[((size_t)(h * L_ + l)) * VF_ + v] * ls[l + 1];
    Weff[((size_t)(i * H_ + h)) * VF_ + v] = (_Float16)(acc * scale);
}

__global__ __launch_bounds__(256)
void pk_beff(const float* __restrict__ bm, const float* __restrict__ logsig,
             const float* __restrict__ intervals, float* __restrict__ beff) {
    int idx = blockIdx.x * blockDim.x + threadIdx.x;
    if (idx >= NSEG * H_) return;
    int h = idx & (H_ - 1);
    int i = idx >> 7;
    float scale = 1.0f / (intervals[i + 1] - intervals[i]);
    const float* ls = logsig + i * 63;
    float acc = 0.0f;
    for (int l = 0; l < L_; ++l)
        acc += bm[h * L_ + l] * ls[l + 1];
    beff[idx] = acc * scale;
}

__global__ __launch_bounds__(256)
void cvt_f16(const float* __restrict__ src, _Float16* __restrict__ dst, int n) {
    int i = blockIdx.x * blockDim.x + threadIdx.x;
    if (i < n) dst[i] = (_Float16)src[i];
}

// y0 = x0 @ W_in.T + b_in  (K = 5: scalar kernel)
__global__ __launch_bounds__(256)
void init_y(const float* __restrict__ x0, const float* __restrict__ Win,
            const float* __restrict__ bin, float* __restrict__ y,
            _Float16* __restrict__ yh) {
    int idx = blockIdx.x * blockDim.x + threadIdx.x;
    if (idx >= B_ * H_) return;
    int b = idx >> 7, h = idx & (H_ - 1);
    float acc = bin[h];
#pragma unroll
    for (int d = 0; d < 5; ++d) acc += x0[b * 5 + d] * Win[h * 5 + d];
    y[idx]  = acc;
    yh[idx] = (_Float16)acc;
}

// ---------------- WMMA GEMM + activation:  out(f16) = act(A @ W^T + bias) ----------------
// A: 512 x K f16 row-major.  W: NTOT x K f16 row-major (i.e. B^T).
// One wave computes a 16x32 output strip (A fragment reused across 2 WMMAs).
template<int NTOT, int ACT>
__global__ __launch_bounds__(128)
void gemm_act(const _Float16* __restrict__ A, const _Float16* __restrict__ W,
              const float* __restrict__ bias, _Float16* __restrict__ out, int K) {
    constexpr int NSTRIP = NTOT / 32;
    int wid    = blockIdx.x * 4 + (threadIdx.x >> 5);
    int mtile  = wid / NSTRIP;
    int nstrip = wid - mtile * NSTRIP;
    int m0 = mtile << 4;
    int n0 = nstrip << 5;
    int lane = threadIdx.x & 31;

    const _Float16* pa  = A + (size_t)m0 * K;
    const _Float16* pb0 = W + (size_t)n0 * K;
    const _Float16* pb1 = W + (size_t)(n0 + 16) * K;

    v8f c0 = {}; v8f c1 = {};
    for (int k0 = 0; k0 < K; k0 += 32) {
        if (k0 + 32 < K) {  // stream next B chunk through global_prefetch
            __builtin_prefetch(pb0 + (size_t)(lane & 15) * K + k0 + 32, 0, 1);
        }
        v16h a  = load_frag(pa  + k0, K);
        v16h b0 = load_frag(pb0 + k0, K);
        v16h b1 = load_frag(pb1 + k0, K);
        c0 = __builtin_amdgcn_wmma_f32_16x16x32_f16(false, a, false, b0, (short)0, c0, false, false);
        c1 = __builtin_amdgcn_wmma_f32_16x16x32_f16(false, a, false, b1, (short)0, c1, false, false);
    }

    // C/D layout: VGPR r -> M = r (+8 for lanes>=16), N = lane&15.
    int rb = (lane >> 4) << 3;
    int nn = lane & 15;
#pragma unroll
    for (int r = 0; r < 8; ++r) {
        int m = m0 + rb + r;
        float v0 = c0[r] + bias[n0 + nn];
        float v1 = c1[r] + bias[n0 + 16 + nn];
        if (ACT == 0) { v0 = fmaxf(v0, 0.0f); v1 = fmaxf(v1, 0.0f); }
        else          { v0 = tanhf(v0);       v1 = tanhf(v1); }
        out[(size_t)m * NTOT + n0 + nn]      = (_Float16)v0;
        out[(size_t)m * NTOT + n0 + 16 + nn] = (_Float16)v1;
    }
}

// ---------------- WMMA GEMM for the vector field + fused Heun update ----------------
// dydt = h2 @ W_eff[seg]^T + b_eff[seg]   (denominator already folded in)
// is_k2 == 0:  k1 = dydt;            ytmp16 = f16(y + dt*k1)
// is_k2 == 1:  y  = y + dt/2*(k1+dydt);  y16 = f16(y)
__global__ __launch_bounds__(128)
void gemm_field(const _Float16* __restrict__ H2, const _Float16* __restrict__ Weff,
                const float* __restrict__ beff, const float* __restrict__ ts,
                const float* __restrict__ intervals,
                const float* yin, float* k1buf, float* yout32,
                _Float16* __restrict__ yout16, int step, int is_k2) {
    // t = t0 + step*dt (+dt for k2) computed exactly as the reference does.
    float t0 = ts[0], t1 = ts[32];
    float dt = (t1 - t0) * (1.0f / (float)NSTEPS);
    float t  = t0 + (float)step * dt + (is_k2 ? dt : 0.0f);
    // searchsorted(intervals[1:], t, 'left') -> count of elements < t
    int j = 0;
#pragma unroll
    for (int s = 1; s <= 32; ++s) j += (intervals[s] < t) ? 1 : 0;
    int seg = j < (NSEG - 1) ? j : (NSEG - 1);

    const _Float16* W    = Weff + (size_t)seg * H_ * VF_;
    const float*    bias = beff + seg * H_;

    constexpr int NSTRIP = H_ / 32;      // 4
    int wid    = blockIdx.x * 4 + (threadIdx.x >> 5);
    int mtile  = wid / NSTRIP;
    int nstrip = wid - mtile * NSTRIP;
    int m0 = mtile << 4;
    int n0 = nstrip << 5;
    int lane = threadIdx.x & 31;

    const _Float16* pa  = H2 + (size_t)m0 * VF_;
    const _Float16* pb0 = W  + (size_t)n0 * VF_;
    const _Float16* pb1 = W  + (size_t)(n0 + 16) * VF_;

    v8f c0 = {}; v8f c1 = {};
    for (int k0 = 0; k0 < VF_; k0 += 32) {
        v16h a  = load_frag(pa  + k0, VF_);
        v16h b0 = load_frag(pb0 + k0, VF_);
        v16h b1 = load_frag(pb1 + k0, VF_);
        c0 = __builtin_amdgcn_wmma_f32_16x16x32_f16(false, a, false, b0, (short)0, c0, false, false);
        c1 = __builtin_amdgcn_wmma_f32_16x16x32_f16(false, a, false, b1, (short)0, c1, false, false);
    }

    int rb = (lane >> 4) << 3;
    int nn = lane & 15;
#pragma unroll
    for (int r = 0; r < 8; ++r) {
        int m = m0 + rb + r;
#pragma unroll
        for (int q = 0; q < 2; ++q) {
            int n = n0 + 16 * q + nn;
            float dydt = (q ? c1[r] : c0[r]) + bias[n];
            size_t o = (size_t)m * H_ + n;
            if (!is_k2) {
                k1buf[o]  = dydt;
                yout16[o] = (_Float16)(yin[o] + dt * dydt);
            } else {
                float yn = yin[o] + 0.5f * dt * (k1buf[o] + dydt);
                yout32[o] = yn;
                yout16[o] = (_Float16)yn;
            }
        }
    }
}

// ---------------- classifier head + softmax (tiny: K=128, 10 classes) ----------------
__global__ __launch_bounds__(128)
void head_softmax(const float* __restrict__ y, const float* __restrict__ Wout,
                  const float* __restrict__ bout, float* __restrict__ out) {
    int b = blockIdx.x * blockDim.x + threadIdx.x;
    if (b >= B_) return;
    float logit[10];
    float mx = -3.4e38f;
#pragma unroll
    for (int c = 0; c < 10; ++c) {
        float a = bout[c];
        for (int k = 0; k < H_; ++k) a += y[(size_t)b * H_ + k] * Wout[c * H_ + k];
        logit[c] = a;
        mx = fmaxf(mx, a);
    }
    float s = 0.0f;
#pragma unroll
    for (int c = 0; c < 10; ++c) { logit[c] = __expf(logit[c] - mx); s += logit[c]; }
    float inv = 1.0f / s;
#pragma unroll
    for (int c = 0; c < 10; ++c) out[(size_t)b * 10 + c] = logit[c] * inv;
}

// ---------------- host orchestration ----------------

static inline size_t alignup(size_t x) { return (x + 255) & ~(size_t)255; }

extern "C" void kernel_launch(void* const* d_in, const int* in_sizes, int n_in,
                              void* d_out, int out_size, void* d_ws, size_t ws_size,
                              hipStream_t stream) {
    (void)in_sizes; (void)n_in; (void)out_size; (void)ws_size;
    const float* ts        = (const float*)d_in[0];   // 33
    const float* logsig    = (const float*)d_in[1];   // 32 x 63
    const float* x0        = (const float*)d_in[2];   // 512 x 5
    const float* intervals = (const float*)d_in[3];   // 33
    const float* W_vf1     = (const float*)d_in[4];   // 256 x 128
    const float* b_vf1     = (const float*)d_in[5];   // 256
    const float* W_vf2     = (const float*)d_in[6];   // 256 x 256
    const float* b_vf2     = (const float*)d_in[7];   // 256
    const float* W_m       = (const float*)d_in[8];   // 7936 x 256
    const float* b_m       = (const float*)d_in[9];   // 7936
    const float* W_in      = (const float*)d_in[10];  // 128 x 5
    const float* b_in      = (const float*)d_in[11];  // 128
    const float* W_out     = (const float*)d_in[12];  // 10 x 128
    const float* b_out     = (const float*)d_in[13];  // 10
    float* out = (float*)d_out;

    // workspace carve
    char* ws = (char*)d_ws;
    size_t off = 0;
    auto carve = [&](size_t bytes) { char* p = ws + off; off += alignup(bytes); return p; };
    _Float16* Weff   = (_Float16*)carve((size_t)NSEG * H_ * VF_ * 2);
    float*    beff   = (float*)   carve((size_t)NSEG * H_ * 4);
    _Float16* Wvf1h  = (_Float16*)carve((size_t)VF_ * H_ * 2);
    _Float16* Wvf2h  = (_Float16*)carve((size_t)VF_ * VF_ * 2);
    float*    y      = (float*)   carve((size_t)B_ * H_ * 4);
    float*    k1     = (float*)   carve((size_t)B_ * H_ * 4);
    _Float16* yh     = (_Float16*)carve((size_t)B_ * H_ * 2);
    _Float16* ytmph  = (_Float16*)carve((size_t)B_ * H_ * 2);
    _Float16* h1     = (_Float16*)carve((size_t)B_ * VF_ * 2);
    _Float16* h2     = (_Float16*)carve((size_t)B_ * VF_ * 2);

    // one-time precompute
    pk_weff<<<NSEG * H_, 256, 0, stream>>>(W_m, logsig, intervals, Weff);
    pk_beff<<<(NSEG * H_ + 255) / 256, 256, 0, stream>>>(b_m, logsig, intervals, beff);
    cvt_f16<<<(VF_ * H_ + 255) / 256, 256, 0, stream>>>(W_vf1, Wvf1h, VF_ * H_);
    cvt_f16<<<(VF_ * VF_ + 255) / 256, 256, 0, stream>>>(W_vf2, Wvf2h, VF_ * VF_);
    init_y<<<(B_ * H_ + 255) / 256, 256, 0, stream>>>(x0, W_in, b_in, y, yh);

    // Heun integration: 20 steps x 2 vector-field evals, each = 3 WMMA GEMMs.
    // grids: gemm_act<256>: (512/16)*(256/32)=256 waves -> 64 blocks of 4 waves
    //        gemm_field   : (512/16)*(128/32)=128 waves -> 32 blocks
    for (int i = 0; i < NSTEPS; ++i) {
        // k1 = vf(t, y)
        gemm_act<VF_, 0><<<64, 128, 0, stream>>>(yh, Wvf1h, b_vf1, h1, H_);
        gemm_act<VF_, 1><<<64, 128, 0, stream>>>(h1, Wvf2h, b_vf2, h2, VF_);
        gemm_field<<<32, 128, 0, stream>>>(h2, Weff, beff, ts, intervals,
                                           y, k1, nullptr, ytmph, i, 0);
        // k2 = vf(t+dt, y + dt*k1); y += dt/2*(k1+k2)
        gemm_act<VF_, 0><<<64, 128, 0, stream>>>(ytmph, Wvf1h, b_vf1, h1, H_);
        gemm_act<VF_, 1><<<64, 128, 0, stream>>>(h1, Wvf2h, b_vf2, h2, VF_);
        gemm_field<<<32, 128, 0, stream>>>(h2, Weff, beff, ts, intervals,
                                           y, k1, y, yh, i, 1);
    }

    head_softmax<<<(B_ + 127) / 128, 128, 0, stream>>>(y, W_out, b_out, out);
}
